// DilatedCNN_86517821215731
// MI455X (gfx1250) — matrix operand order repaired and checked by reading, not compiled, not run on hardware
//
#include <hip/hip_runtime.h>
#include <hip/hip_bf16.h>

typedef __attribute__((ext_vector_type(16))) __bf16 v16bf;
typedef __attribute__((ext_vector_type(8)))  float  v8f;
typedef unsigned short u16;
typedef unsigned int   u32;

#define NROWS 8192
#define DIM   1024
#define K3    3072

// round-to-nearest-even fp32 -> bf16 (bit pattern)
__device__ __forceinline__ u16 f2bf(float f) {
    u32 u = __float_as_uint(f);
    u32 r = (u + 0x7FFFu + ((u >> 16) & 1u)) >> 16;
    return (u16)r;
}

union Frag32 { uint4 q[2]; v16bf v; };
union Pack8  { uint4 q; u16 s[8]; };

__device__ __forceinline__ v16bf ldFrag(const u16* base, int elemOff) {
    Frag32 f;
    f.q[0] = *(const uint4*)(base + elemOff);
    f.q[1] = *(const uint4*)(base + elemOff + 8);
    return f.v;
}

// ---------------- fp32 -> bf16 bulk convert (8 elems / thread) ----------------
__global__ __launch_bounds__(256) void cvt_bf16(const float* __restrict__ src,
                                                u16* __restrict__ dst, int n) {
    int i = (blockIdx.x * 256 + threadIdx.x) * 8;
    if (i + 8 > n) return;
    float4 a = *(const float4*)(src + i);
    float4 b = *(const float4*)(src + i + 4);
    Pack8 p;
    p.s[0] = f2bf(a.x); p.s[1] = f2bf(a.y); p.s[2] = f2bf(a.z); p.s[3] = f2bf(a.w);
    p.s[4] = f2bf(b.x); p.s[5] = f2bf(b.y); p.s[6] = f2bf(b.z); p.s[7] = f2bf(b.w);
    *(uint4*)(dst + i) = p.q;
}

// ------- weights: fp32 [l][k=3072][n=1024] -> bf16 [l][n=1024][k=3072] -------
__global__ __launch_bounds__(256) void wcvt_transpose(const float* __restrict__ W,
                                                      u16* __restrict__ Wt) {
    __shared__ u16 t[32][33];
    const int tx = threadIdx.x & 31, ty = threadIdx.x >> 5;   // 32x8
    const int n0 = blockIdx.x * 32, k0 = blockIdx.y * 32, l = blockIdx.z;
    const float* src = W + ((size_t)l * K3 + k0) * DIM + n0;
    #pragma unroll
    for (int j = 0; j < 4; ++j)
        t[ty + j * 8][tx] = f2bf(src[(size_t)(ty + j * 8) * DIM + tx]);
    __syncthreads();
    u16* dst = Wt + ((size_t)l * DIM + n0) * K3 + k0;
    #pragma unroll
    for (int j = 0; j < 4; ++j)
        dst[(size_t)(ty + j * 8) * K3 + tx] = t[tx][ty + j * 8];
}

// ----------------------------- fused layer GEMM ------------------------------
// out[i] = 0.5*Xold[i] + 0.5*relu( Xb[i]@W0 + Xb[i-d]@W1 + Xb[i+d]@W2 + bias )
// 128x128 tile / WG, 8 waves (4x2), 32x64 per wave, K chunks of 32.
// Double-buffered LDS with COMPILE-TIME buffer roles (even/odd phases) so the
// accumulator tuples stay pinned across the loop; one barrier per chunk.
__global__ __launch_bounds__(256) void dilated_layer(
    const u16* __restrict__ Xb,    // [8192][1024] bf16 (pre-layer snapshot)
    const u16* __restrict__ Wt,    // [1024][3072] bf16, n-major (this layer)
    const u16* __restrict__ oobb,  // [1024] bf16
    const float* Xold,             // fp32 pre-layer values (may alias out)
    const float* __restrict__ bias,// [1024] fp32
    float* out,                    // fp32 (may alias Xold)
    int delta)
{
    __shared__ u16 lA[2][128 * 32];  // A chunk, fragment-permuted K groups
    __shared__ u16 lB[2][128 * 32];  // B chunk, [n][k]
    const int tid  = threadIdx.x;
    const int lane = tid & 31, wave = tid >> 5;
    const int wm = wave & 3, wn = wave >> 2;        // 4x2 wave grid
    const int lm = lane & 15, hi = lane >> 4;
    const int tileM = blockIdx.y * 128;
    const int tileN = blockIdx.x * 128;

    // --- constant per-thread staging task decode (2 tasks of 8 bf16 each) ---
    const int rowA0 = tid >> 2,         gA0 = tid & 3;
    const int rowA1 = (tid + 256) >> 2, gA1 = tid & 3;   // id1 = tid+256: same g
    const int pgA0 = ((gA0 & 1) << 1) | (gA0 >> 1);  // K groups -> [0-7,16-23,8-15,24-31]
    const int pgA1 = ((gA1 & 1) << 1) | (gA1 >> 1);
    const int sA0 = rowA0 * 32 + pgA0 * 8;           // LDS store offsets (elems)
    const int sA1 = rowA1 * 32 + pgA1 * 8;
    const int sB0 = rowA0 * 32 + pgA0 * 8;           // B: n = rowA, g permuted like A
    const int sB1 = rowA1 * 32 + pgA1 * 8;

    // --- constant fragment read offsets (elems) ---
    const int aOff0 = (wm * 32 + lm) * 32 + hi * 16;
    const int aOff1 = aOff0 + 16 * 32;
    const int bOff  = (wn * 64 + lm) * 32 + hi * 16; // + ns*16*32

    v8f acc[2][4] = {};
    uint4 av0, av1, bv0, bv1;                 // staging registers
    const u16 *aP0, *aP1, *bP0, *bP1;         // per-blk source base pointers

    auto loadRegs = [&](int kchunk) {
        const int ko = kchunk * 32;           // elems; 64B per chunk
        av0 = *(const uint4*)(aP0 + ko); av1 = *(const uint4*)(aP1 + ko);
        bv0 = *(const uint4*)(bP0 + ko); bv1 = *(const uint4*)(bP1 + ko);
    };
    auto storeRegs = [&](int buf) {           // buf is always a literal -> static addrs
        *(uint4*)&lA[buf][sA0] = av0;  *(uint4*)&lA[buf][sA1] = av1;
        *(uint4*)&lB[buf][sB0] = bv0;  *(uint4*)&lB[buf][sB1] = bv1;
    };
    auto compute = [&](int buf) {             // buf is always a literal
        const u16* A = &lA[buf][0];
        const u16* B = &lB[buf][0];
        v16bf aF[2], bF[4];
        aF[0] = ldFrag(A, aOff0);
        aF[1] = ldFrag(A, aOff1);
        #pragma unroll
        for (int ns = 0; ns < 4; ++ns) bF[ns] = ldFrag(B, bOff + ns * 16 * 32);
        #pragma unroll
        for (int ms = 0; ms < 2; ++ms)
            #pragma unroll
            for (int ns = 0; ns < 4; ++ns)
                acc[ms][ns] = __builtin_amdgcn_wmma_f32_16x16x32_bf16(
                    false, aF[ms], false, bF[ns],
                    (short)0, acc[ms][ns], false, false);
    };

    for (int blk = 0; blk < 3; ++blk) {
        // --- hoisted per-blk source pointers (validity fixed per row/blk) ---
        {
            int s0 = tileM + rowA0, s1 = tileM + rowA1;
            bool v0 = true, v1 = true;
            if (blk == 1)      { s0 -= delta; s1 -= delta; v0 = (s0 >= 0);    v1 = (s1 >= 0); }
            else if (blk == 2) { s0 += delta; s1 += delta; v0 = (s0 < NROWS); v1 = (s1 < NROWS); }
            aP0 = v0 ? (Xb + (size_t)s0 * DIM + gA0 * 8) : (oobb + gA0 * 8);
            aP1 = v1 ? (Xb + (size_t)s1 * DIM + gA1 * 8) : (oobb + gA1 * 8);
        }
        bP0 = Wt + (size_t)(tileN + rowA0) * K3 + blk * DIM + gA0 * 8;
        bP1 = Wt + (size_t)(tileN + rowA1) * K3 + blk * DIM + gA1 * 8;

        // --- prime chunk 0 -> buf0 ---
        loadRegs(0);
        storeRegs(0);
        __syncthreads();

        // --- steady state: even chunk from buf0 / odd chunk from buf1 ---
        #pragma unroll 1
        for (int i = 0; i < 15; ++i) {
            loadRegs(2 * i + 1); compute(0); storeRegs(1); __syncthreads();
            loadRegs(2 * i + 2); compute(1); storeRegs(0); __syncthreads();
        }
        loadRegs(31); compute(0); storeRegs(1); __syncthreads();
        compute(1);
        __syncthreads();       // protect buffers before next blk's prime
    }

    // ---- epilogue: bias + relu + residual blend, fp32 store ----
    #pragma unroll
    for (int ms = 0; ms < 2; ++ms) {
        #pragma unroll
        for (int ns = 0; ns < 4; ++ns) {
            const int col = tileN + wn * 64 + ns * 16 + lm;
            const float b = bias[col];
            #pragma unroll
            for (int v = 0; v < 8; ++v) {
                const int row = tileM + wm * 32 + ms * 16 + v + hi * 8;
                float y = fmaxf(acc[ms][ns][v] + b, 0.0f);
                size_t off = (size_t)row * DIM + col;
                out[off] = 0.5f * Xold[off] + 0.5f * y;
            }
        }
    }
}

extern "C" void kernel_launch(void* const* d_in, const int* in_sizes, int n_in,
                              void* d_out, int out_size, void* d_ws, size_t ws_size,
                              hipStream_t stream) {
    const float* X   = (const float*)d_in[0];   // [8192][1024]
    const float* Ws  = (const float*)d_in[1];   // [4][3072][1024]
    const float* bs  = (const float*)d_in[2];   // [4][1024]
    const float* oob = (const float*)d_in[3];   // [1024]
    float* out = (float*)d_out;                 // [8192][1024]

    u16* wt = (u16*)d_ws;                       // 4*1024*3072 bf16 (24 MB)
    u16* xb = wt + (size_t)4 * DIM * K3;        // 8192*1024 bf16 (16 MB)
    u16* ob = xb + (size_t)NROWS * DIM;         // 1024 bf16

    // one-shot (re-run each call, deterministic): weights transpose+convert, oob convert
    wcvt_transpose<<<dim3(32, 96, 4), 256, 0, stream>>>(Ws, wt);
    cvt_bf16<<<dim3(1), 256, 0, stream>>>(oob, ob, DIM);

    const int deltas[4] = {1, 2, 4, 1};
    for (int l = 0; l < 4; ++l) {
        const float* xf = (l == 0) ? X : out;
        cvt_bf16<<<dim3(NROWS * DIM / (256 * 8)), 256, 0, stream>>>(xf, xb, NROWS * DIM);
        dilated_layer<<<dim3(8, 64), 256, 0, stream>>>(
            xb, wt + (size_t)l * DIM * K3, ob, xf, bs + l * DIM, out, deltas[l]);
    }
}
